// MultisourcePerceiverEncoder_23441931501608
// MI455X (gfx1250) — matrix operand order repaired and checked by reading, not compiled, not run on hardware
//
#include <hip/hip_runtime.h>
#include <cstdint>
#include <cstddef>

// ---------------------------------------------------------------------------
// MultisourcePerceiverEncoder on gfx1250 (MI455X), bf16 WMMA pipeline.
//   B=4, M=16384, N=512 latents, H=8 heads, D=64, key-dim per head = 128
// ---------------------------------------------------------------------------

#define B_    4
#define M_    16384
#define N_    512
#define H_    8
#define D_    64
#define VDIM  256
#define CDIM  128
#define INNER 512
#define KV2   1024   // 2*INNER columns of the kv projections

typedef __attribute__((ext_vector_type(16))) __bf16 v16bf;
typedef __attribute__((ext_vector_type(8)))  float  v8f;
typedef __attribute__((ext_vector_type(4)))  unsigned int v4u;
typedef __attribute__((ext_vector_type(8)))  int    v8i;
typedef __attribute__((ext_vector_type(4)))  int    v4i;
typedef unsigned short ushort_t;

#if defined(__gfx1250__) && __has_builtin(__builtin_amdgcn_tensor_load_to_lds) && \
    __has_builtin(__builtin_amdgcn_s_wait_tensorcnt)
#define USE_TDM 1
#endif

union V16BF { v16bf v; uint4 u4[2]; };

__device__ __forceinline__ unsigned short f2bf(float f) {
  unsigned int u = __builtin_bit_cast(unsigned int, f);
  unsigned int r = (u + 0x7FFFu + ((u >> 16) & 1u)) >> 16;  // RNE
  return (unsigned short)r;
}

// B-operand (32x16 bf16): lane = column, 16 contiguous K values (K = half*16 + j)
__device__ __forceinline__ v16bf ld_b16(const ushort_t* p) {
  V16BF t;
  t.u4[0] = *(const uint4*)(p);
  t.u4[1] = *(const uint4*)(p + 8);
  return t.v;
}
// A-operand (16x32 bf16): lane = row, K(j) = kf0 + half*16 + j (contiguous)
__device__ __forceinline__ v16bf ld_a16(const ushort_t* rowbase, int kf0, int hf) {
  V16BF t;
  t.u4[0] = *(const uint4*)(rowbase + kf0 + hf * 16);
  t.u4[1] = *(const uint4*)(rowbase + kf0 + hf * 16 + 8);
  return t.v;
}

__device__ __forceinline__ float red16(float s) {   // sum within 16-lane group
  for (int m = 1; m < 16; m <<= 1) s += __shfl_xor(s, m, 16);
  return s;
}
__device__ __forceinline__ float redmax16(float s) {
  for (int m = 1; m < 16; m <<= 1) s = fmaxf(s, __shfl_xor(s, m, 16));
  return s;
}

// ---------------------------------------------------------------------------
// Kernel 0: pack fp32 weights -> bf16 WMMA B-operand tiles.
// ---------------------------------------------------------------------------
__global__ void k_pack(const float* __restrict__ Wkvv,
                       const float* __restrict__ Wkvc,
                       ushort_t* __restrict__ pkv,
                       ushort_t* __restrict__ pkc) {
  const int lane = threadIdx.x & 31;
  const int ln = lane & 15, hf = lane >> 4;
  int bid = blockIdx.x;
  const float* src;
  ushort_t* dst;
  int kt, nt, tile;
  if (bid < 512) { src = Wkvv; dst = pkv; kt = bid >> 6; nt = bid & 63; tile = bid; }
  else { bid -= 512; src = Wkvc; dst = pkc; kt = bid >> 6; nt = bid & 63; tile = bid; }
  const int n = nt * 16 + ln;
  ushort_t* out = dst + (size_t)tile * 512 + lane * 16;
#pragma unroll
  for (int j = 0; j < 16; ++j) {
    const int k = kt * 32 + hf * 16 + j;
    out[j] = f2bf(src[(size_t)k * KV2 + n]);
  }
}

// ---------------------------------------------------------------------------
// Kernel 1: latent query projections + RMSNorm -> qcat bf16 [H][N][128]
// ---------------------------------------------------------------------------
__global__ void k_qproj(const float* __restrict__ lv, const float* __restrict__ lc,
                        const float* __restrict__ Wqv, const float* __restrict__ bqv,
                        const float* __restrict__ gqv,
                        const float* __restrict__ Wqc, const float* __restrict__ bqc,
                        const float* __restrict__ gqc,
                        ushort_t* __restrict__ qcat) {
  __shared__ float ssq;
  const int n = blockIdx.x, tid = threadIdx.x;
  const int c0 = tid, c1 = tid + 256;
  // ----- values path: fan-in 256 -----
  float a0 = bqv[c0], a1 = bqv[c1];
  for (int k = 0; k < VDIM; ++k) {
    const float x = lv[(size_t)n * VDIM + k];
    a0 += x * Wqv[(size_t)k * INNER + c0];
    a1 += x * Wqv[(size_t)k * INNER + c1];
  }
  if (tid == 0) ssq = 0.f;
  __syncthreads();
  atomicAdd(&ssq, a0 * a0 + a1 * a1);
  __syncthreads();
  {
    const float rs = rsqrtf(ssq * (1.0f / INNER) + 1e-6f);
    const float v0 = a0 * rs * gqv[c0], v1 = a1 * rs * gqv[c1];
    qcat[(((size_t)(c0 >> 6) * N_) + n) * 128 + (c0 & 63)] = f2bf(v0);
    qcat[(((size_t)(c1 >> 6) * N_) + n) * 128 + (c1 & 63)] = f2bf(v1);
  }
  __syncthreads();
  // ----- coords path: fan-in 128 -----
  a0 = bqc[c0]; a1 = bqc[c1];
  for (int k = 0; k < CDIM; ++k) {
    const float x = lc[(size_t)n * CDIM + k];
    a0 += x * Wqc[(size_t)k * INNER + c0];
    a1 += x * Wqc[(size_t)k * INNER + c1];
  }
  if (tid == 0) ssq = 0.f;
  __syncthreads();
  atomicAdd(&ssq, a0 * a0 + a1 * a1);
  __syncthreads();
  {
    const float rs = rsqrtf(ssq * (1.0f / INNER) + 1e-6f);
    const float v0 = a0 * rs * gqc[c0], v1 = a1 * rs * gqc[c1];
    qcat[(((size_t)(c0 >> 6) * N_) + n) * 128 + 64 + (c0 & 63)] = f2bf(v0);
    qcat[(((size_t)(c1 >> 6) * N_) + n) * 128 + 64 + (c1 & 63)] = f2bf(v1);
  }
}

// ---------------------------------------------------------------------------
// Kernel 2: KV projections + RMSNorm via WMMA.
//   K output:  kcat [b, h, m, 128]   (kv half in 0:64, kc half in 64:128)
//   V outputs: vv/vc TRANSPOSED [b, h, d, M]  (so attention stages contiguously)
// ---------------------------------------------------------------------------
__device__ __forceinline__ void kv_phase(
    const ushort_t* Alds, int fanin,
    const ushort_t* __restrict__ pack,
    const float* __restrict__ bias, const float* __restrict__ g,
    float* rowsq, int which,
    ushort_t* __restrict__ kcat, ushort_t* __restrict__ vout,
    int b, int m0, int wave, int lane) {
  const int ln = lane & 15, hf = lane >> 4;
  v8f acc[8] = {};
  const ushort_t* arow = Alds + ln * fanin;
  const int ksteps = fanin >> 5;
  for (int ks = 0; ks < ksteps; ++ks) {
    const v16bf A = ld_a16(arow, ks * 32, hf);
#pragma unroll
    for (int t = 0; t < 8; ++t) {
      const int tile = ks * 64 + wave * 8 + t;
      const v16bf Bm = ld_b16(pack + (size_t)tile * 512 + lane * 16);
      acc[t] = __builtin_amdgcn_wmma_f32_16x16x32_bf16(
          false, A, false, Bm, (short)0, acc[t], false, false);
    }
  }
#pragma unroll
  for (int t = 0; t < 8; ++t) {
    const float bv = bias[wave * 128 + t * 16 + ln];
#pragma unroll
    for (int i = 0; i < 8; ++i) acc[t][i] += bv;
  }
#pragma unroll
  for (int i = 0; i < 8; ++i) {
    float s = 0.f;
#pragma unroll
    for (int t = 0; t < 8; ++t) s += acc[t][i] * acc[t][i];
    s = red16(s);
    if (ln == 0) atomicAdd(&rowsq[i + 8 * hf], s);
  }
  __syncthreads();
#pragma unroll
  for (int i = 0; i < 8; ++i) {
    const int m = m0 + i + 8 * hf;
    const float rs = rsqrtf(rowsq[i + 8 * hf] * (1.0f / KV2) + 1e-6f);
#pragma unroll
    for (int t = 0; t < 8; ++t) {
      const int col = wave * 128 + t * 16 + ln;
      const float val = acc[t][i] * rs * g[col];
      if (col < INNER) {                                  // key half
        const int head = col >> 6, d = col & 63;
        kcat[((((size_t)b * H_ + head) * M_) + m) * 128 + d + (which ? 64 : 0)] =
            f2bf(val);
      } else {                                            // value half (transposed)
        const int cc = col - INNER, head = cc >> 6, d = cc & 63;
        vout[(((size_t)b * H_ + head) * 64 + d) * M_ + m] = f2bf(val);
      }
    }
  }
}

__global__ void k_kvproj(const float* __restrict__ V, const float* __restrict__ C,
                         const ushort_t* __restrict__ pkv,
                         const ushort_t* __restrict__ pkc,
                         const float* __restrict__ bkvv, const float* __restrict__ gkvv,
                         const float* __restrict__ bkvc, const float* __restrict__ gkvc,
                         ushort_t* __restrict__ kcat,
                         ushort_t* __restrict__ vv, ushort_t* __restrict__ vc) {
  __shared__ ushort_t Av[16 * VDIM];
  __shared__ ushort_t Ac[16 * CDIM];
  __shared__ float rowsq[16];
  const int tid = threadIdx.x;
  const int wave = tid >> 5, lane = tid & 31;
  const int b = blockIdx.x / (M_ / 16);
  const int m0 = (blockIdx.x % (M_ / 16)) * 16;

  for (int i = tid; i < 16 * VDIM; i += 256) {
    const int r = i / VDIM, k = i % VDIM;
    Av[i] = f2bf(V[(((size_t)b * M_) + m0 + r) * VDIM + k]);
  }
  for (int i = tid; i < 16 * CDIM; i += 256) {
    const int r = i / CDIM, k = i % CDIM;
    Ac[i] = f2bf(C[(((size_t)b * M_) + m0 + r) * CDIM + k]);
  }
  if (tid < 16) rowsq[tid] = 0.f;
  __syncthreads();

  kv_phase(Av, VDIM, pkv, bkvv, gkvv, rowsq, 0, kcat, vv, b, m0, wave, lane);
  __syncthreads();
  if (tid < 16) rowsq[tid] = 0.f;
  __syncthreads();
  kv_phase(Ac, CDIM, pkc, bkvc, gkvc, rowsq, 1, kcat, vc, b, m0, wave, lane);
}

// ---------------------------------------------------------------------------
// Kernel 3: flash-attention. Block = (b, h, 64 latents); 4 waves x 16 latents.
// K chunk staged by the Tensor Data Mover (TDM) when available; V chunks are
// contiguous uint4 copies (already transposed in global memory).
// ---------------------------------------------------------------------------
__global__ void k_attn(const ushort_t* __restrict__ qcat,
                       const ushort_t* __restrict__ kcat,
                       const ushort_t* __restrict__ vv,
                       const ushort_t* __restrict__ vc,
                       float* __restrict__ attn_out) {
  __shared__ ushort_t Kch[64 * 128];        // 16 KB  K chunk   [m][128]
  __shared__ ushort_t VvT[64 * 80];         // 10 KB  Vv chunk  [d][m] (padded)
  __shared__ ushort_t VcT[64 * 80];         // 10 KB  Vc chunk  [d][m]
  __shared__ ushort_t Pst[4 * 16 * 64];     //  8 KB  per-wave P staging [16][64]

  const int tid = threadIdx.x;
  const int wave = tid >> 5, lane = tid & 31;
  const int ln = lane & 15, hf = lane >> 4;
  const int bid = blockIdx.x;
  const int b = bid / (H_ * 8);
  const int h = (bid / 8) % H_;
  const int nt = bid & 7;
  const int lat0 = nt * 64 + wave * 16;

  // preload this wave's Q tile (16 latents x 128) as 4 A-operands
  v16bf q[4];
  {
    const ushort_t* qrow = qcat + (((size_t)h * N_) + lat0 + ln) * 128;
#pragma unroll
    for (int s = 0; s < 4; ++s) q[s] = ld_a16(qrow, s * 32, hf);
  }

  float Mi[8], Li[8];
#pragma unroll
  for (int i = 0; i < 8; ++i) { Mi[i] = -1e30f; Li[i] = 0.f; }
  v8f ov[4] = {}, oc[4] = {};

  const size_t kbase = (((size_t)b * H_ + h) * M_) * 128;
  const size_t vbase = (((size_t)b * H_ + h) * 64) * M_;
#ifdef USE_TDM
  const unsigned ldsK = (unsigned)(size_t)(const void*)Kch;
#endif

  for (int mi = 0; mi < M_; mi += 64) {
    __syncthreads();
    // ---- stage K chunk (contiguous 16 KB) ----
#ifdef USE_TDM
    if (wave == 0) {
      const unsigned long long ga =
          (unsigned long long)(size_t)(kcat + kbase + (size_t)mi * 128);
      // D# group0: count=1 | lds_addr | global_addr[56:0] | type=2
      v4u g0 = {1u, ldsK, (unsigned)ga,
                (unsigned)((ga >> 32) & 0x01FFFFFFull) | 0x80000000u};
      // D# group1: data_size=8B, tensor_dim0/1 huge (no OOB clip),
      //            tile_dim0=2048 (16 KB), tile_dim1=1, stride large.
      v8i g1 = {0x00030000, (int)0xFFFF0000, (int)0xFFFF7FFF, 0x08007FFF,
                1, 0x7FFFFFFF, 0, 0};
      v4i z4 = {0, 0, 0, 0};
#if __clang_major__ >= 23
      v8i z8 = {0, 0, 0, 0, 0, 0, 0, 0};
      __builtin_amdgcn_tensor_load_to_lds(g0, g1, z4, z4, z8, 0);
#else
      __builtin_amdgcn_tensor_load_to_lds(g0, g1, z4, z4, 0);
#endif
    }
#else
    {
      const uint4* gk = (const uint4*)(kcat + kbase + (size_t)mi * 128);
      for (int i = tid; i < 1024; i += 128) ((uint4*)Kch)[i] = gk[i];
      __builtin_prefetch((const void*)(gk + 1024), 0, 1);
    }
#endif
    // ---- stage V chunks [d][m] (global already transposed; contiguous) ----
    for (int i = tid; i < 512; i += 128) {
      const int d = i >> 3, mloc = (i & 7) * 8;
      *(uint4*)&VvT[d * 80 + mloc] =
          *(const uint4*)(vv + vbase + (size_t)d * M_ + mi + mloc);
      *(uint4*)&VcT[d * 80 + mloc] =
          *(const uint4*)(vc + vbase + (size_t)d * M_ + mi + mloc);
    }
#ifdef USE_TDM
    if (wave == 0) __builtin_amdgcn_s_wait_tensorcnt(0);
#endif
    __syncthreads();

    // ---- scores: S[t] = Q (16x128) . K^T tile (128 x 16m), t = m-subtile ----
    v8f S[4];
#pragma unroll
    for (int t = 0; t < 4; ++t) {
      v8f a = {};
      const ushort_t* krow = Kch + (t * 16 + ln) * 128;
#pragma unroll
      for (int s = 0; s < 4; ++s) {
        const v16bf Bm = ld_b16(krow + s * 32 + hf * 16);
        a = __builtin_amdgcn_wmma_f32_16x16x32_bf16(
            false, q[s], false, Bm, (short)0, a, false, false);
      }
#pragma unroll
      for (int i = 0; i < 8; ++i) a[i] *= 0.125f;   // 1/sqrt(64)
      S[t] = a;
    }

    // ---- online softmax update ----
#pragma unroll
    for (int i = 0; i < 8; ++i) {
      float r = S[0][i];
      r = fmaxf(r, S[1][i]); r = fmaxf(r, S[2][i]); r = fmaxf(r, S[3][i]);
      r = redmax16(r);
      const float nm = fmaxf(Mi[i], r);
      const float alpha = __expf(Mi[i] - nm);
      Mi[i] = nm;
      Li[i] *= alpha;
#pragma unroll
      for (int t = 0; t < 4; ++t) { ov[t][i] *= alpha; oc[t][i] *= alpha; }
      float ps = 0.f;
#pragma unroll
      for (int t = 0; t < 4; ++t) {
        const float p = __expf(S[t][i] - nm);
        S[t][i] = p;
        ps += p;
      }
      Li[i] += red16(ps);
    }

    // ---- P: C-layout -> bf16 LDS [16][64] (wave-private), reload as A ----
    ushort_t* P = Pst + wave * (16 * 64);
#pragma unroll
    for (int t = 0; t < 4; ++t)
#pragma unroll
      for (int i = 0; i < 8; ++i)
        P[(i + 8 * hf) * 64 + t * 16 + ln] = f2bf(S[t][i]);

#pragma unroll
    for (int ms = 0; ms < 2; ++ms) {                 // two 32-token reductions
      const v16bf Ap = ld_a16(P + ln * 64, ms * 32, hf);
#pragma unroll
      for (int dt = 0; dt < 4; ++dt) {
        const ushort_t* vrow = VvT + (dt * 16 + ln) * 80 + ms * 32 + hf * 16;
        const ushort_t* crow = VcT + (dt * 16 + ln) * 80 + ms * 32 + hf * 16;
        ov[dt] = __builtin_amdgcn_wmma_f32_16x16x32_bf16(
            false, Ap, false, ld_b16(vrow), (short)0, ov[dt], false, false);
        oc[dt] = __builtin_amdgcn_wmma_f32_16x16x32_bf16(
            false, Ap, false, ld_b16(crow), (short)0, oc[dt], false, false);
      }
    }
  }

  // ---- epilogue: normalize by Li, write merged-head fp32 [B][N][1024] ----
#pragma unroll
  for (int i = 0; i < 8; ++i) {
    const int n = lat0 + i + 8 * hf;
    const float inv = 1.0f / Li[i];
#pragma unroll
    for (int dt = 0; dt < 4; ++dt) {
      const int col = h * 64 + dt * 16 + ln;
      attn_out[(((size_t)b * N_) + n) * 1024 + col] = ov[dt][i] * inv;
      attn_out[(((size_t)b * N_) + n) * 1024 + 512 + col] = oc[dt][i] * inv;
    }
  }
}

// ---------------------------------------------------------------------------
// Kernel 4: output projections (fp32). One block per (b, n) row.
// ---------------------------------------------------------------------------
__global__ void k_oproj(const float* __restrict__ attn_out,
                        const float* __restrict__ Wov, const float* __restrict__ bov,
                        const float* __restrict__ Woc, const float* __restrict__ boc,
                        float* __restrict__ out) {
  __shared__ float rowv[INNER];
  __shared__ float rowc[INNER];
  const int tid = threadIdx.x;
  const int b = blockIdx.x / N_, n = blockIdx.x % N_;
  const float* src = attn_out + (((size_t)b * N_) + n) * 1024;
  for (int i = tid; i < INNER; i += 256) { rowv[i] = src[i]; rowc[i] = src[INNER + i]; }
  __syncthreads();
  {
    float a = bov[tid];
    for (int k = 0; k < INNER; ++k) a += rowv[k] * Wov[(size_t)k * VDIM + tid];
    out[(((size_t)b * N_) + n) * VDIM + tid] = a;
  }
  if (tid < CDIM) {
    float a = boc[tid];
    for (int k = 0; k < INNER; ++k) a += rowc[k] * Woc[(size_t)k * CDIM + tid];
    out[(size_t)B_ * N_ * VDIM + (((size_t)b * N_) + n) * CDIM + tid] = a;
  }
}

// ---------------------------------------------------------------------------
// Host-side launcher
// ---------------------------------------------------------------------------
extern "C" void kernel_launch(void* const* d_in, const int* in_sizes, int n_in,
                              void* d_out, int out_size, void* d_ws, size_t ws_size,
                              hipStream_t stream) {
  (void)in_sizes; (void)n_in; (void)out_size; (void)ws_size;
  const float* V   = (const float*)d_in[0];
  const float* C   = (const float*)d_in[1];
  const float* lv  = (const float*)d_in[2];
  const float* lc  = (const float*)d_in[3];
  const float* Wqv = (const float*)d_in[4];
  const float* bqv = (const float*)d_in[5];
  const float* gqv = (const float*)d_in[6];
  const float* Wqc = (const float*)d_in[7];
  const float* bqc = (const float*)d_in[8];
  const float* gqc = (const float*)d_in[9];
  const float* Wkvv = (const float*)d_in[10];
  const float* bkvv = (const float*)d_in[11];
  const float* gkvv = (const float*)d_in[12];
  const float* Wkvc = (const float*)d_in[13];
  const float* bkvc = (const float*)d_in[14];
  const float* gkvc = (const float*)d_in[15];
  const float* Wov = (const float*)d_in[16];
  const float* bov = (const float*)d_in[17];
  const float* Woc = (const float*)d_in[18];
  const float* boc = (const float*)d_in[19];

  char* ws = (char*)d_ws;
  const size_t SZ_QCAT = (size_t)H_ * N_ * 128 * 2;        //   1 MB
  const size_t SZ_KCAT = (size_t)B_ * H_ * M_ * 128 * 2;   // 128 MB
  const size_t SZ_V    = (size_t)B_ * H_ * M_ * 64 * 2;    //  64 MB each
  const size_t SZ_ATTN = (size_t)B_ * N_ * 1024 * 4;       //   8 MB
  const size_t SZ_PKV  = (size_t)512 * 512 * 2;            // 512 KB
  size_t off = 0;
  ushort_t* qcat = (ushort_t*)(ws + off); off += SZ_QCAT;
  ushort_t* kcat = (ushort_t*)(ws + off); off += SZ_KCAT;
  ushort_t* vv   = (ushort_t*)(ws + off); off += SZ_V;
  ushort_t* vc   = (ushort_t*)(ws + off); off += SZ_V;
  float*    attn = (float*)(ws + off);    off += SZ_ATTN;
  ushort_t* pkv  = (ushort_t*)(ws + off); off += SZ_PKV;
  ushort_t* pkc  = (ushort_t*)(ws + off);

  k_pack<<<dim3(768), dim3(32), 0, stream>>>(Wkvv, Wkvc, pkv, pkc);
  k_qproj<<<dim3(N_), dim3(256), 0, stream>>>(lv, lc, Wqv, bqv, gqv,
                                              Wqc, bqc, gqc, qcat);
  k_kvproj<<<dim3(B_ * M_ / 16), dim3(256), 0, stream>>>(
      V, C, pkv, pkc, bkvv, gkvv, bkvc, gkvc, kcat, vv, vc);
  k_attn<<<dim3(B_ * H_ * 8), dim3(128), 0, stream>>>(qcat, kcat, vv, vc, attn);
  k_oproj<<<dim3(B_ * N_), dim3(256), 0, stream>>>(attn, Wov, bov, Woc, boc,
                                                   (float*)d_out);
}
